// RoutingLayer_29360396436009
// MI455X (gfx1250) — matrix operand order repaired and checked by reading, not compiled, not run on hardware
//
#include <hip/hip_runtime.h>

// MI455X (gfx1250, wave32) capsule-routing kernel.
// Roofline: ~1.3 GFLOP vs ~55 MB HBM (gathers hit 192MB L2) -> bandwidth bound.
// The per-node k-batched GEMVs are expressed as real GEMMs so the routing math
// runs on v_wmma_f32_16x16x32_f16:
//   P  (20x8) = Z(20x128)   x blockdiag(u)(128x8)   [2 Mtiles x 4 Ksteps]
//   U  (8x16) = diag-blocks of Z^T(128x20) x P(20x8) [8 Mtiles x 1 Kstep]
// f16 operands are safe: all values are unit-normalized capsules / probabilities.

typedef __attribute__((ext_vector_type(16))) _Float16 v16h;
typedef __attribute__((ext_vector_type(8)))  _Float16 v8h;
typedef __attribute__((ext_vector_type(4)))  _Float16 v4h;
typedef __attribute__((ext_vector_type(8)))  float    v8f;
typedef __attribute__((ext_vector_type(4)))  float    v4f;

#define D_DIM   128
#define M_NB    20
#define ZSTRIDE 136   // halves per LDS row: 272B, 16B-aligned, 4-bank skew/row
#define WAVES_B 4

// ---------------- Kernel A: per-capsule L2 normalize, write f32 + f16 tables
__global__ void caps_norm_kernel(const float* __restrict__ in,
                                 float* __restrict__ xf32,
                                 _Float16* __restrict__ xf16, int n) {
  int row  = (int)((blockIdx.x * blockDim.x + threadIdx.x) >> 5);
  int lane = threadIdx.x & 31;
  if (row > n) return;
  size_t off = (size_t)row * D_DIM + (size_t)lane * 4;
  if (row == n) {                      // zero pad row (gather target n)
    v4f z = {};  *(v4f*)(xf32 + off) = z;
    v4h hz = {}; *(v4h*)(xf16 + off) = hz;
    return;
  }
  v4f v = *(const v4f*)(in + off);
  // lane l owns elements 4l..4l+3 -> capsule k = l>>2; reduce over 4-lane group
  float ss = v.x * v.x + v.y * v.y + v.z * v.z + v.w * v.w;
  ss += __shfl_xor(ss, 1);
  ss += __shfl_xor(ss, 2);
  float sc = 1.0f / fmaxf(sqrtf(ss), 1e-12f);
  v.x *= sc; v.y *= sc; v.z *= sc; v.w *= sc;
  *(v4f*)(xf32 + off) = v;
  v4h h;
  h[0] = (_Float16)v.x; h[1] = (_Float16)v.y;
  h[2] = (_Float16)v.z; h[3] = (_Float16)v.w;
  *(v4h*)(xf16 + off) = h;
}

// ---------------- Kernel B: one wave per node, WMMA routing iterations
__global__ void __launch_bounds__(WAVES_B * 32)
caps_route_kernel(const int* __restrict__ neighbors,
                  const int* __restrict__ max_iter_p,
                  const float* __restrict__ xf32,
                  const _Float16* __restrict__ xf16,
                  float* __restrict__ out, int n) {
  __shared__ __align__(16) _Float16 z_lds[WAVES_B][32][ZSTRIDE]; // Z, rows 20..31 zero
  __shared__ __align__(16) float    u_f[WAVES_B][128];           // raw u accumulators
  __shared__ __align__(16) _Float16 u_h[WAVES_B][128];           // normalized u (f16)
  __shared__ __align__(16) _Float16 p_h[WAVES_B][32][8];         // softmaxed p (f16)

  const int  lane   = threadIdx.x & 31;
  const int  w      = threadIdx.x >> 5;
  const int  node   = (int)blockIdx.x * WAVES_B + w;
  const bool active = node < n;
  const int  mi     = *max_iter_p;
  const bool hi     = (lane & 16) != 0;
  const int  l15    = lane & 15;
  const int  base   = hi ? 8 : 0;      // K-base for f16 A-fragment lane halves

  // this node's normalized x (lane l owns elements 4l..4l+3)
  v4f xr = {};
  if (active) xr = *(const v4f*)(xf32 + (size_t)node * D_DIM + (size_t)lane * 4);

  // zero pad rows of Z (m = 20..31); one v4h chunk per lane per row
  v4h hz = {};
#pragma unroll
  for (int m = M_NB; m < 32; ++m)
    *(v4h*)&z_lds[w][m][lane * 4] = hz;

  // gather 20 neighbor rows of the f16 table (wave cooperative, 8B/lane)
  if (active) {
#pragma unroll
    for (int m = 0; m < M_NB; ++m) {
      int idx = neighbors[node * M_NB + m] + node * 5;   // reference index math
      v4h hv = *(const v4h*)(xf16 + (size_t)idx * D_DIM + (size_t)lane * 4);
      *(v4h*)&z_lds[w][m][lane * 4] = hv;
    }
  } else {
#pragma unroll
    for (int m = 0; m < M_NB; ++m)
      *(v4h*)&z_lds[w][m][lane * 4] = hz;
  }
  __syncthreads();

  for (int it = 0; it < mi; ++it) {
    // ---- B operand for the u-update: P padded to 32x16 (cols 8..15 not used)
    v16h bp = {};
    if (it == 0) {
      // softmax(0) over k=8 -> uniform 1/8 (pad rows killed by zero Z columns)
#pragma unroll
      for (int e = 0; e < 16; ++e) bp[e] = (_Float16)0.125f;
    } else {
      // ---- matmul 1: logits L = Z(32x128) x blockdiag(u)(128x16)
      v8f L0 = {}, L1 = {};
#pragma unroll
      for (int kk = 0; kk < 4; ++kk) {
        // Bdiag K-slab kk: lane holds col n = l15; row j nonzero iff n == j>>4
        v16h bd = {};
        _Float16 h0 = (_Float16)0.0f;
#pragma unroll
        for (int e = 0; e < 8; ++e) {
          bd[e]     = (l15 == 2 * kk)     ? u_h[w][kk * 32 + base + e] : h0;
          bd[8 + e] = (l15 == 2 * kk + 1) ? u_h[w][kk * 32 + 16 + base + e] : h0;
        }
        {
          // A tile mt2=0 (rows 0..15): two 16B chunks per lane from its Z row
          int r = l15;
          v8h lo  = *(const v8h*)&z_lds[w][r][kk * 32 + base];
          v8h hi8 = *(const v8h*)&z_lds[w][r][kk * 32 + 16 + base];
          v16h a = __builtin_shufflevector(lo, hi8, 0, 1, 2, 3, 4, 5, 6, 7,
                                           8, 9, 10, 11, 12, 13, 14, 15);
          L0 = __builtin_amdgcn_wmma_f32_16x16x32_f16(false, a, false, bd,
                                                      (short)0, L0, false, false);
        }
        {
          // A tile mt2=1 (rows 16..31; rows >=20 are zero)
          int r = 16 + l15;
          v8h lo  = *(const v8h*)&z_lds[w][r][kk * 32 + base];
          v8h hi8 = *(const v8h*)&z_lds[w][r][kk * 32 + 16 + base];
          v16h a = __builtin_shufflevector(lo, hi8, 0, 1, 2, 3, 4, 5, 6, 7,
                                           8, 9, 10, 11, 12, 13, 14, 15);
          L1 = __builtin_amdgcn_wmma_f32_16x16x32_f16(false, a, false, bd,
                                                      (short)0, L1, false, false);
        }
      }
      // ---- softmax over k (8 valid lanes per row) and transpose via LDS
#pragma unroll
      for (int mt2 = 0; mt2 < 2; ++mt2) {
        v8f Lv = mt2 ? L1 : L0;
#pragma unroll
        for (int v = 0; v < 8; ++v) {
          float lv = Lv[v];
          float mx = lv;
          mx = fmaxf(mx, __shfl_xor(mx, 1));
          mx = fmaxf(mx, __shfl_xor(mx, 2));
          mx = fmaxf(mx, __shfl_xor(mx, 4));
          float ev = __expf(lv - mx);
          float sv = ev;
          sv += __shfl_xor(sv, 1);
          sv += __shfl_xor(sv, 2);
          sv += __shfl_xor(sv, 4);
          float pv = ev / sv;
          if (l15 < 8) {            // valid capsule columns only
            int m = mt2 * 16 + v + (hi ? 8 : 0);
            p_h[w][m][l15] = (_Float16)pv;
          }
        }
      }
      __syncthreads();
      // B-fragment: lane holds P column n=l15, K rows base+e and 16+base+e
#pragma unroll
      for (int e = 0; e < 8; ++e) {
        bp[e]     = p_h[w][base + e][l15];
        bp[8 + e] = p_h[w][16 + base + e][l15];
      }
    }

    // ---- matmul 2: tile mt gives Uall rows mt*16..mt*16+15; useful col = mt
#pragma unroll
    for (int mt = 0; mt < 8; ++mt) {
      int c = mt * 16 + l15;              // Z^T row = Z column
      v16h a2 = {};
#pragma unroll
      for (int e = 0; e < 8; ++e) a2[e]     = z_lds[w][base + e][c];       // K=base+e
#pragma unroll
      for (int e = 0; e < 8; ++e) a2[8 + e] = z_lds[w][16 + base + e][c];  // K=16+base+e
      v8f zc = {};
      v8f d = __builtin_amdgcn_wmma_f32_16x16x32_f16(false, a2, false, bp,
                                                     (short)0, zc, false, false);
      if (l15 == mt) {                    // extract column mt -> u_k raw sums
        int bj = mt * 16 + (hi ? 8 : 0);
#pragma unroll
        for (int v = 0; v < 8; ++v) u_f[w][bj + v] = d[v];
      }
    }
    __syncthreads();

    // ---- u = sum + x; normalize per capsule except on final iteration
    v4f uv = *(const v4f*)&u_f[w][lane * 4];
    uv.x += xr.x; uv.y += xr.y; uv.z += xr.z; uv.w += xr.w;
    if (it == mi - 1) {
      if (active)
        *(v4f*)(out + (size_t)node * D_DIM + (size_t)lane * 4) = uv;
    } else {
      float ss = uv.x * uv.x + uv.y * uv.y + uv.z * uv.z + uv.w * uv.w;
      ss += __shfl_xor(ss, 1);
      ss += __shfl_xor(ss, 2);
      float sc = 1.0f / fmaxf(sqrtf(ss), 1e-12f);
      v4h hv;
      hv[0] = (_Float16)(uv.x * sc); hv[1] = (_Float16)(uv.y * sc);
      hv[2] = (_Float16)(uv.z * sc); hv[3] = (_Float16)(uv.w * sc);
      *(v4h*)&u_h[w][lane * 4] = hv;
    }
    __syncthreads();
  }
}

extern "C" void kernel_launch(void* const* d_in, const int* in_sizes, int n_in,
                              void* d_out, int out_size, void* d_ws, size_t ws_size,
                              hipStream_t stream) {
  const float* input     = (const float*)d_in[0];
  const int*   neighbors = (const int*)d_in[1];
  const int*   max_iter  = (const int*)d_in[2];
  float*       out       = (float*)d_out;

  const int n = in_sizes[0] / D_DIM;   // B*N nodes

  // Workspace: (n+1) x 128 f32 normalized table, then (n+1) x 128 f16 copy.
  float*    xf32 = (float*)d_ws;
  _Float16* xf16 = (_Float16*)((char*)d_ws + (size_t)(n + 1) * D_DIM * sizeof(float));

  const int rowsA  = n + 1;                 // includes zero pad row
  const int gridA  = (rowsA + 7) / 8;       // 256 threads = 8 waves = 8 rows
  caps_norm_kernel<<<gridA, 256, 0, stream>>>(input, xf32, xf16, n);

  const int gridB = (n + WAVES_B - 1) / WAVES_B;
  caps_route_kernel<<<gridB, WAVES_B * 32, 0, stream>>>(neighbors, max_iter,
                                                        xf32, xf16, out, n);
}
